// SeparateHiddenGCVAEEncoder_16286515987223
// MI455X (gfx1250) — matrix operand order repaired and checked by reading, not compiled, not run on hardware
//
#include <hip/hip_runtime.h>
#include <hip/hip_bf16.h>
#include <math.h>

#define NNODES 50000
#define NEDGES 800000
#define FEATD  128
#define CONDD  64
#define HIDD   128
#define LATD   64

typedef __attribute__((ext_vector_type(16))) __bf16 v16bf;
typedef __attribute__((ext_vector_type(8)))  float  v8f;

// ---------- small utility kernels ----------
__global__ void k_f32_to_bf16(const float* __restrict__ x,
                              __bf16* __restrict__ y, long n) {
    long i = (long)blockIdx.x * blockDim.x + threadIdx.x;
    if (i < n) y[i] = (__bf16)x[i];
}

// Fuse W_mean [128,64] and W_logvar [128,64] into fp32 Wml [128,128]
__global__ void k_fuse_ml(const float* __restrict__ Wm, const float* __restrict__ Wl,
                          float* __restrict__ Wml) {
    int i = blockIdx.x * blockDim.x + threadIdx.x;
    if (i >= HIDD * 2 * LATD) return;
    int k = i >> 7, j = i & 127;
    Wml[i] = (j < LATD) ? Wm[k * LATD + j] : Wl[k * LATD + (j - LATD)];
}

// Pack fp32 B[K,Nc] into WMMA-B fragment-major bf16 layout:
// Bp[(((tile*KS + ks)*32 + lane)*16 + i]  where lane={hi,n}, i={v,j}:
//   value = B[(ks*32 + 16*hi + 2*v + j) * Nc + tile*16 + n]
// so each lane's 16 operand values are 32 contiguous bytes (two b128 loads).
__global__ void k_pack_B(const float* __restrict__ B, __bf16* __restrict__ Bp,
                         int K, int Nc) {
    int o = blockIdx.x * blockDim.x + threadIdx.x;
    if (o >= K * Nc) return;
    int i    = o & 15;
    int lane = (o >> 4) & 31;
    int rest = o >> 9;                 // tile*KS + ks
    int KS   = K >> 5;
    int ks   = rest % KS;
    int tile = rest / KS;
    int hi = lane >> 4, n = lane & 15;
    int v = i >> 1, j = i & 1;
    int k = ks * 32 + 16 * hi + 2 * v + j;
    int col = tile * 16 + n;
    Bp[o] = (__bf16)B[k * Nc + col];
}

__global__ void k_deg_init(float* __restrict__ deg) {
    int i = blockIdx.x * blockDim.x + threadIdx.x;
    if (i < NNODES) deg[i] = 1.0f;  // self-loop
}
__global__ void k_deg_acc(const int* __restrict__ dst, float* __restrict__ deg) {
    int e = blockIdx.x * blockDim.x + threadIdx.x;
    if (e < NEDGES) atomicAdd(&deg[dst[e]], 1.0f);
}
__global__ void k_dinv(float* __restrict__ deg) {
    int i = blockIdx.x * blockDim.x + threadIdx.x;
    if (i < NNODES) deg[i] = rsqrtf(deg[i]);  // deg >= 1 always
}

// ---------- bf16 WMMA GEMM ----------
// C[:, coff + 0..Nc) = A[M,K] @ B[K,Nc].  One 16(M) x 64(N) strip per wave:
// one A fragment feeds 4 v_wmma_f32_16x16x32_bf16 into 4 accumulators.
// B is pre-packed fragment-major (see k_pack_B) -> all operand loads are b128.
template<int K>
__global__ __launch_bounds__(256)
void k_wmma_gemm_bf16(const __bf16* __restrict__ A,   // [M,K] row-major bf16
                      const __bf16* __restrict__ Bp,  // packed bf16 fragments
                      __bf16* __restrict__ C,         // [M,ldc] bf16
                      int M, int Nc, int ldc, int coff)
{
    constexpr int KS = K / 32;
    const int wave = threadIdx.x >> 5;
    const int lane = threadIdx.x & 31;
    const int tilesN4 = Nc >> 6;               // Nc multiple of 64
    const int numWT = (M >> 4) * tilesN4;
    const int wt = blockIdx.x * 8 + wave;      // uniform across the wave
    if (wt >= numWT) return;                   // whole-wave exit -> EXEC all 1s

    const int tm  = wt / tilesN4;
    const int tn0 = (wt - tm * tilesN4) * 4;
    const int m16 = lane & 15;
    const int hi  = lane >> 4;
    const int arow = tm * 16 + m16;

    const uint4* Abase = (const uint4*)(A + (size_t)arow * K);
    const uint4* Bbase = (const uint4*)Bp;

    v8f acc0 = {}, acc1 = {}, acc2 = {}, acc3 = {};
    #pragma unroll
    for (int ks = 0; ks < KS; ++ks) {
        union { v16bf v; uint4 q[2]; } Af, B0, B1, B2, B3;
        Af.q[0] = Abase[ks * 4 + 2 * hi];
        Af.q[1] = Abase[ks * 4 + 2 * hi + 1];
        size_t b0 = ((size_t)((tn0 + 0) * KS + ks) * 32 + lane) * 2;
        size_t b1 = ((size_t)((tn0 + 1) * KS + ks) * 32 + lane) * 2;
        size_t b2 = ((size_t)((tn0 + 2) * KS + ks) * 32 + lane) * 2;
        size_t b3 = ((size_t)((tn0 + 3) * KS + ks) * 32 + lane) * 2;
        B0.q[0] = Bbase[b0]; B0.q[1] = Bbase[b0 + 1];
        B1.q[0] = Bbase[b1]; B1.q[1] = Bbase[b1 + 1];
        B2.q[0] = Bbase[b2]; B2.q[1] = Bbase[b2 + 1];
        B3.q[0] = Bbase[b3]; B3.q[1] = Bbase[b3 + 1];
        acc0 = __builtin_amdgcn_wmma_f32_16x16x32_bf16(false, Af.v, false, B0.v, (short)0, acc0, false, false);
        acc1 = __builtin_amdgcn_wmma_f32_16x16x32_bf16(false, Af.v, false, B1.v, (short)0, acc1, false, false);
        acc2 = __builtin_amdgcn_wmma_f32_16x16x32_bf16(false, Af.v, false, B2.v, (short)0, acc2, false, false);
        acc3 = __builtin_amdgcn_wmma_f32_16x16x32_bf16(false, Af.v, false, B3.v, (short)0, acc3, false, false);
    }

    // C layout: reg r -> row (tm*16 + 8*hi + r), col tile*16 + m16
    #pragma unroll
    for (int r = 0; r < 8; ++r) {
        __bf16* crow = C + (size_t)(tm * 16 + 8 * hi + r) * ldc + coff;
        crow[(tn0 + 0) * 16 + m16] = (__bf16)acc0[r];
        crow[(tn0 + 1) * 16 + m16] = (__bf16)acc1[r];
        crow[(tn0 + 2) * 16 + m16] = (__bf16)acc2[r];
        crow[(tn0 + 3) * 16 + m16] = (__bf16)acc3[r];
    }
}

// ---------- edge aggregation: AGG[dst] += H[src] * (dinv[src]*dinv[dst]) ----------
// One thread per (edge, 8-column chunk); 16B bf16 gather (L2-resident), fp32 atomic scatter.
__global__ __launch_bounds__(256)
void k_agg(const int* __restrict__ src, const int* __restrict__ dst,
           const float* __restrict__ dinv, const __bf16* __restrict__ H,
           float* __restrict__ AGG, int width)
{
    long t = (long)blockIdx.x * blockDim.x + threadIdx.x;
    int chunks = width >> 3;
    long e = t / chunks;
    int  c = (int)(t - e * chunks);
    if (e >= NEDGES) return;
    int s = src[e], d = dst[e];
    float nrm = dinv[s] * dinv[d];
    union { uint4 q; __bf16 b[8]; } v;
    v.q = *(const uint4*)(H + (size_t)s * width + c * 8);
    float* out = AGG + (size_t)d * width + c * 8;
    #pragma unroll
    for (int j = 0; j < 8; ++j)
        atomicAdd(out + j, (float)v.b[j] * nrm);
}

// ---------- epilogue: fold self-loop + bias + tanh, emit bf16 ----------
__global__ void k_act(const float* __restrict__ AGG, const __bf16* __restrict__ H,
                      const float* __restrict__ b0, const float* __restrict__ b1, int split,
                      const float* __restrict__ dinv, __bf16* __restrict__ Out,
                      int width, long total)
{
    long t = (long)blockIdx.x * blockDim.x + threadIdx.x;
    if (t >= total) return;
    long i = t / width;
    int  j = (int)(t - i * width);
    float sl = dinv[i]; sl *= sl;                       // self-loop norm
    float bias = (j < split) ? b0[j] : b1[j - split];
    float v = AGG[t] + (float)H[t] * sl + bias;
    Out[t] = (__bf16)tanhf(v);
}

// ---------- final: mean/logvar epilogue + reparameterization ----------
__global__ void k_final(const float* __restrict__ AGG, const __bf16* __restrict__ ML,
                        const float* __restrict__ bm, const float* __restrict__ bl,
                        const float* __restrict__ dinv, const float* __restrict__ noise,
                        float* __restrict__ out)
{
    int t = blockIdx.x * blockDim.x + threadIdx.x;
    if (t >= NNODES * LATD) return;
    int i = t / LATD, j = t - (t / LATD) * LATD;
    float sl = dinv[i] * dinv[i];
    float m  = AGG[(size_t)i * 128 + j]      + (float)ML[(size_t)i * 128 + j]      * sl + bm[j];
    float lv = AGG[(size_t)i * 128 + 64 + j] + (float)ML[(size_t)i * 128 + 64 + j] * sl + bl[j];
    float z  = noise[t] * __expf(0.5f * lv) + m;
    const size_t stride = (size_t)NNODES * LATD;
    out[t]              = z;
    out[stride + t]     = m;
    out[2 * stride + t] = lv;
}

extern "C" void kernel_launch(void* const* d_in, const int* in_sizes, int n_in,
                              void* d_out, int out_size, void* d_ws, size_t ws_size,
                              hipStream_t stream)
{
    (void)in_sizes; (void)n_in; (void)out_size; (void)ws_size;
    const float* feature   = (const float*)d_in[0];
    const float* condition = (const float*)d_in[1];
    const int*   edge      = (const int*)d_in[2];
    const float* noise     = (const float*)d_in[3];
    const float* W_f2h     = (const float*)d_in[4];
    const float* b_f2h     = (const float*)d_in[5];
    const float* W_c2h     = (const float*)d_in[6];
    const float* b_c2h     = (const float*)d_in[7];
    const float* W_h2h     = (const float*)d_in[8];
    const float* b_h2h     = (const float*)d_in[9];
    const float* W_mean    = (const float*)d_in[10];
    const float* b_mean    = (const float*)d_in[11];
    const float* W_logvar  = (const float*)d_in[12];
    const float* b_logvar  = (const float*)d_in[13];
    const int* src = edge;
    const int* dst = edge + NEDGES;

    // ---- workspace carve-out (256B aligned) ----
    char* ws = (char*)d_ws;
    size_t off = 0;
    auto carve = [&](size_t bytes) {
        void* p = ws + off;
        off = (off + bytes + 255) & ~(size_t)255;
        return p;
    };
    float*  dinv  = (float*) carve((size_t)NNODES * 4);
    float*  WmlF  = (float*) carve((size_t)HIDD * 2 * LATD * 4);        // fp32 fused heads
    __bf16* Wf    = (__bf16*)carve((size_t)FEATD * HIDD * 2);           // packed
    __bf16* Wc    = (__bf16*)carve((size_t)CONDD * HIDD * 2);           // packed
    __bf16* Wh    = (__bf16*)carve((size_t)2 * HIDD * HIDD * 2);        // packed
    __bf16* Wml   = (__bf16*)carve((size_t)HIDD * 2 * LATD * 2);        // packed
    __bf16* Af    = (__bf16*)carve((size_t)NNODES * FEATD * 2);         // later: H3 bf16 [N,128]
    __bf16* Ac    = (__bf16*)carve((size_t)NNODES * CONDD * 2);
    __bf16* HFC   = (__bf16*)carve((size_t)NNODES * 256 * 2);           // later: H2 bf16 [N,128]
    float*  AGG   = (float*) carve((size_t)NNODES * 256 * 4);           // reused 3x
    __bf16* H2in  = (__bf16*)carve((size_t)NNODES * 256 * 2);           // later: ML bf16 [N,128]

    const int T = 256;
    auto blk = [](long n, int t) { return (int)((n + t - 1) / t); };

    // ---- weights: fuse heads, pack all B matrices into fragment layout ----
    k_fuse_ml<<<blk((long)HIDD * 2 * LATD, T), T, 0, stream>>>(W_mean, W_logvar, WmlF);
    k_pack_B<<<blk((long)FEATD * HIDD, T), T, 0, stream>>>(W_f2h, Wf, FEATD, HIDD);
    k_pack_B<<<blk((long)CONDD * HIDD, T), T, 0, stream>>>(W_c2h, Wc, CONDD, HIDD);
    k_pack_B<<<blk((long)2 * HIDD * HIDD, T), T, 0, stream>>>(W_h2h, Wh, 2 * HIDD, HIDD);
    k_pack_B<<<blk((long)HIDD * 2 * LATD, T), T, 0, stream>>>(WmlF, Wml, HIDD, 2 * LATD);

    // ---- activations to bf16 ----
    k_f32_to_bf16<<<blk((long)NNODES * FEATD, T), T, 0, stream>>>(feature, Af, (long)NNODES * FEATD);
    k_f32_to_bf16<<<blk((long)NNODES * CONDD, T), T, 0, stream>>>(condition, Ac, (long)NNODES * CONDD);

    // ---- symmetric normalization: dinv = rsqrt(1 + indegree) ----
    k_deg_init<<<blk(NNODES, T), T, 0, stream>>>(dinv);
    k_deg_acc<<<blk(NEDGES, T), T, 0, stream>>>(dst, dinv);
    k_dinv<<<blk(NNODES, T), T, 0, stream>>>(dinv);

    const int gemmBlocks = ((NNODES / 16) * (HIDD / 64) + 7) / 8;   // 782

    // ---- layer 1: HFC[:,0:128) = feat@Wf ; HFC[:,128:256) = cond@Wc ----
    k_wmma_gemm_bf16<FEATD><<<gemmBlocks, T, 0, stream>>>(Af, Wf, HFC, NNODES, HIDD, 256, 0);
    k_wmma_gemm_bf16<CONDD><<<gemmBlocks, T, 0, stream>>>(Ac, Wc, HFC, NNODES, HIDD, 256, 128);
    hipMemsetAsync(AGG, 0, (size_t)NNODES * 256 * 4, stream);
    k_agg<<<blk((long)NEDGES * 32, T), T, 0, stream>>>(src, dst, dinv, HFC, AGG, 256);
    k_act<<<blk((long)NNODES * 256, T), T, 0, stream>>>(AGG, HFC, b_f2h, b_c2h, 128, dinv,
                                                        H2in, 256, (long)NNODES * 256);

    // ---- layer 2: H2 = tanh(agg(H2in @ Wh)) ----
    k_wmma_gemm_bf16<2 * HIDD><<<gemmBlocks, T, 0, stream>>>(H2in, Wh, HFC, NNODES, HIDD, 128, 0);
    hipMemsetAsync(AGG, 0, (size_t)NNODES * 128 * 4, stream);
    k_agg<<<blk((long)NEDGES * 16, T), T, 0, stream>>>(src, dst, dinv, HFC, AGG, 128);
    k_act<<<blk((long)NNODES * 128, T), T, 0, stream>>>(AGG, HFC, b_h2h, b_h2h, 128, dinv,
                                                        Af, 128, (long)NNODES * 128);

    // ---- heads: [mean|logvar] = agg(H3 @ Wml) ----
    k_wmma_gemm_bf16<HIDD><<<gemmBlocks, T, 0, stream>>>(Af, Wml, H2in, NNODES, 2 * LATD, 128, 0);
    hipMemsetAsync(AGG, 0, (size_t)NNODES * 128 * 4, stream);
    k_agg<<<blk((long)NEDGES * 16, T), T, 0, stream>>>(src, dst, dinv, H2in, AGG, 128);

    // ---- reparameterize + write (z, mean, logvar) ----
    k_final<<<blk((long)NNODES * LATD, T), T, 0, stream>>>(AGG, H2in, b_mean, b_logvar,
                                                           dinv, noise, (float*)d_out);
}
